// BondMessagePassing_88914412961905
// MI455X (gfx1250) — matrix operand order repaired and checked by reading, not compiled, not run on hardware
//
#include <hip/hip_runtime.h>

// ---------------------------------------------------------------------------
// BondMessagePassing (D-MPNN) for MI455X / gfx1250, wave32 + WMMA bf16 + TDM.
//   h0 = relu([x[src], edge_attr] @ W_i + b_i)              [E,80]x[80,128]
//   4x: m = scatter(dst,+h) + scatter(src,-h[e^1]); h = relu(h0 + san(m)@W_h + b_h)
//   out = relu([x, san(scatter(dst,+h))] @ W_o + b_o)       [N,192]x[192,128]
// Matrix math on v_wmma_f32_16x16x32_bf16; weights staged to LDS by the
// Tensor Data Mover (tensor_load_to_lds, TENSORcnt); activations f32 in HBM.
// ---------------------------------------------------------------------------

typedef __bf16 bf16_t;
typedef __attribute__((ext_vector_type(16))) __bf16        v16bf;
typedef __attribute__((ext_vector_type(8)))  float         v8f;
typedef __attribute__((ext_vector_type(4)))  unsigned int  u32x4;
typedef __attribute__((ext_vector_type(8)))  int           i32x8;
typedef __attribute__((ext_vector_type(4)))  int           i32x4;

#define HID 128

#if defined(__has_builtin)
#if __has_builtin(__builtin_amdgcn_tensor_load_to_lds)
#define HAVE_TDM 1
#endif
#endif
#ifndef HAVE_TDM
#define HAVE_TDM 0
#endif

__device__ __forceinline__ float sanitize_f(float v) {
    if (__builtin_isnan(v)) return 0.0f;
    if (__builtin_isinf(v)) return v > 0.0f ? 1000.0f : -1000.0f;
    return v;
}

#if HAVE_TDM
// 1-D TDM copy: `bytes` (multiple of 8) from global `gsrc` into LDS at `ldst`.
// D# per ISA 08_async_tensor S8: group0 {count=1, lds_addr, global_addr, type=2},
// group1 {data_size=3(8B), tensor_dim0=tile_dim0=stride=n8, tensor_dim1=1}.
__device__ __forceinline__ void tdm_load_to_lds(const void* gsrc, void* ldst, int bytes) {
    const unsigned lds_off = (unsigned)(unsigned long long)ldst;  // generic->LDS offset (addr[31:0])
    const unsigned long long ga = (unsigned long long)gsrc;
    const unsigned n8 = ((unsigned)bytes) >> 3;
    u32x4 g0 = { 1u,                                   // count=1, user descriptor
                 lds_off,                              // bits 63:32  lds_addr
                 (unsigned)(ga & 0xFFFFFFFFu),         // bits 95:64  global_addr lo
                 (unsigned)((ga >> 32) & 0x01FFFFFFu)  // bits 120:96 global_addr hi
                   | 0x80000000u };                    // bits 127:126 type=2
    i32x8 g1 = { (int)(3u << 16),      // data_size=8B; wg_mask=0; no pad/iter/barrier
                 (int)(n8 << 16),      // tensor_dim0[15:0]
                 (int)(1u << 16),      // tensor_dim0[31:16]=0 | tensor_dim1[15:0]=1
                 (int)(n8 << 16),      // tensor_dim1[31:16]=0 | tile_dim0=n8
                 0,                    // tile_dim1=0 (1-D), tile_dim2=0
                 (int)n8,              // tensor_dim0_stride[31:0]
                 0, 0 };
    i32x4 gz = { 0, 0, 0, 0 };
#if defined(__clang_major__) && (__clang_major__ >= 23)
    i32x8 gz8 = { 0, 0, 0, 0, 0, 0, 0, 0 };
    __builtin_amdgcn_tensor_load_to_lds(g0, g1, gz, gz, gz8, 0);
#else
    __builtin_amdgcn_tensor_load_to_lds(g0, g1, gz, gz, 0);
#endif
}

__device__ __forceinline__ void wait_tensorcnt0() {
#if __has_builtin(__builtin_amdgcn_s_wait_tensorcnt)
    __builtin_amdgcn_s_wait_tensorcnt(0);
#else
    asm volatile("s_wait_tensorcnt 0x0" ::: "memory");
#endif
}
#endif // HAVE_TDM

// Pre-convert f32 weights [K,128] -> bf16 column-major with padded stride KPS:
// out[col*KPS + k] = W[k*128 + col] (k<K), 0 for K<=k<KPS.
__global__ __launch_bounds__(256)
void convert_weights_kernel(const float* __restrict__ W, bf16_t* __restrict__ out,
                            int K, int KPS) {
    const int idx = blockIdx.x * 256 + threadIdx.x;
    if (idx >= 128 * KPS) return;
    const int col = idx / KPS;
    const int k   = idx - col * KPS;
    out[idx] = (bf16_t)((k < K) ? W[k * 128 + col] : 0.0f);
}

// MODE 0: A = [x[src[e]], edge_attr[e]]          (K=80,  KP=96)  -> h0
// MODE 1: A = sanitize(m[n])                     (K=128, KP=128) -> h (adds h0, relu)
// MODE 2: A = [x[n], sanitize(m[n])]             (K=192, KP=192) -> out (relu+sanitize)
template<int K, int KP, int MODE>
__global__ __launch_bounds__(256)
void gemm_wmma_kernel(const float*  __restrict__ A0,      // x (mode 0/2) or m (mode 1)
                      const float*  __restrict__ A1,      // edge_attr (mode 0) or m (mode 2)
                      const int*    __restrict__ edge_index, // mode 0: src = edge_index[e]
                      const bf16_t* __restrict__ Bw,      // weights bf16 col-major [128*KPS]
                      const float*  __restrict__ bias,    // [128]
                      const float*  __restrict__ h0add,   // mode 1: residual h0
                      float* __restrict__ out,            // [nrows,128]
                      int nrows) {
    constexpr int KPS = KP + 8;  // padded LDS row stride (odd multiple of 16B)
    __shared__ __align__(16) bf16_t Al[128 * KPS];  // activations, row-major
    __shared__ __align__(16) bf16_t Bl[128 * KPS];  // weights, column-major

    const int tid  = threadIdx.x;
    const int base = blockIdx.x * 128;

    // ---- B tile: async DMA via Tensor Data Mover (one issue from wave 0) ----
#if HAVE_TDM
    if ((tid >> 5) == 0) {
        tdm_load_to_lds(Bw, Bl, 128 * KPS * (int)sizeof(bf16_t));
    }
#else
    {
        const unsigned* s = (const unsigned*)Bw;
        unsigned* d = (unsigned*)Bl;
        for (int idx = tid; idx < 128 * KPS / 2; idx += 256) d[idx] = s[idx];
    }
#endif

    // ---- A tile: gather + sanitize + f32->bf16 (overlaps the TDM) ----
    for (int idx = tid; idx < 128 * KP; idx += 256) {
        const int r = idx / KP;
        const int c = idx - r * KP;
        const int row = min(base + r, nrows - 1);
        float v = 0.0f;
        if (MODE == 0) {
            const int s = edge_index[row];            // src node of edge `row`
            if (c < 64)       v = A0[s * 64 + c];
            else if (c < 80)  v = A1[row * 16 + (c - 64)];
        } else if (MODE == 1) {
            v = sanitize_f(A0[row * HID + c]);
        } else {
            if (c < 64) v = A0[row * 64 + c];
            else        v = sanitize_f(A1[row * HID + (c - 64)]);
        }
        Al[r * KPS + c] = (bf16_t)v;
    }

#if HAVE_TDM
    if ((tid >> 5) == 0) wait_tensorcnt0();   // TENSORcnt is per-wave (issuing wave)
#endif
    __syncthreads();

    // ---- WMMA: each wave computes 32 rows x 64 cols (2x4 tiles of 16x16) ----
    const int wave = tid >> 5;
    const int lane = tid & 31;
    const int rw   = (wave & 3) * 32;   // row block
    const int cw   = (wave >> 2) * 64;  // col block
    const int lrow = lane & 15;
    const int lhi  = lane >> 4;

    v8f acc[2][4];
#pragma unroll
    for (int rh = 0; rh < 2; ++rh)
#pragma unroll
        for (int t = 0; t < 4; ++t)
            acc[rh][t] = (v8f){0.f,0.f,0.f,0.f,0.f,0.f,0.f,0.f};

#pragma unroll
    for (int ks = 0; ks < KP / 32; ++ks) {
        // A fragment (ISA 7.12.2, 16-bit A 16x32): lane l -> row l&15,
        // halves 0..7 -> K=kb..kb+7, halves 8..15 -> K=kb+16..kb+23, kb=8*(l>>4)
        union { uint4 u[2]; v16bf v; } fa[2];
        const int kb = ks * 32 + (lhi << 3);
#pragma unroll
        for (int rh = 0; rh < 2; ++rh) {
            const int ao = (rw + rh * 16 + lrow) * KPS + kb;
            fa[rh].u[0] = *(const uint4*)&Al[ao];
            fa[rh].u[1] = *(const uint4*)&Al[ao + 16];
        }
        // B fragment (32x16, 16-bit): lane l -> col l&15, same K order as A (kb..kb+7, kb+16..kb+23)
        union { uint4 u[2]; v16bf v; } fb[4];
        const int kbb = ks * 32 + (lhi << 3);
#pragma unroll
        for (int t = 0; t < 4; ++t) {
            const int bo = (cw + t * 16 + lrow) * KPS + kbb;
            fb[t].u[0] = *(const uint4*)&Bl[bo];
            fb[t].u[1] = *(const uint4*)&Bl[bo + 16];
        }
#pragma unroll
        for (int rh = 0; rh < 2; ++rh)
#pragma unroll
            for (int t = 0; t < 4; ++t)
                acc[rh][t] = __builtin_amdgcn_wmma_f32_16x16x32_bf16(
                    false, fa[rh].v, false, fb[t].v, (short)0, acc[rh][t], false, false);
    }

    // ---- epilogue: bias (+ h0) + relu (+ sanitize), store f32 ----
#pragma unroll
    for (int rh = 0; rh < 2; ++rh)
#pragma unroll
        for (int t = 0; t < 4; ++t) {
            const int col = cw + t * 16 + lrow;
            const float bv = bias[col];
#pragma unroll
            for (int g = 0; g < 8; ++g) {
                const int row = base + rw + rh * 16 + g + (lhi << 3);
                if (row < nrows) {
                    float v = acc[rh][t][g] + bv;
                    if (MODE == 1) v += h0add[row * HID + col];
                    v = fmaxf(v, 0.0f);
                    if (MODE == 2) v = sanitize_f(v);
                    out[row * HID + col] = v;
                }
            }
        }
}

// One wave per edge e: m[dst[e]] += h[e]; if withSub: m[src[e^1]] -= h[e].
// (Reference: m[src[i]] -= h[i^1]  <=>  m[src[e^1]] -= h[e] with e = i^1.)
__global__ __launch_bounds__(256)
void scatter_kernel(const float* __restrict__ h,
                    const int*   __restrict__ edge_index,
                    float* __restrict__ m,
                    int E, int withSub) {
    const int e    = blockIdx.x * 8 + (threadIdx.x >> 5);
    const int lane = threadIdx.x & 31;
    if (e >= E) return;
    const float4 hv = *(const float4*)&h[e * HID + lane * 4];
    const int dst = edge_index[E + e];
    float* md = &m[dst * HID + lane * 4];
    atomicAdd(&md[0], hv.x); atomicAdd(&md[1], hv.y);
    atomicAdd(&md[2], hv.z); atomicAdd(&md[3], hv.w);
    if (withSub) {
        const int s2 = edge_index[e ^ 1];   // src of reverse edge
        float* ms = &m[s2 * HID + lane * 4];
        atomicAdd(&ms[0], -hv.x); atomicAdd(&ms[1], -hv.y);
        atomicAdd(&ms[2], -hv.z); atomicAdd(&ms[3], -hv.w);
    }
}

extern "C" void kernel_launch(void* const* d_in, const int* in_sizes, int n_in,
                              void* d_out, int out_size, void* d_ws, size_t ws_size,
                              hipStream_t stream) {
    const float* x          = (const float*)d_in[0];
    const int*   edge_index = (const int*)  d_in[1];
    const float* edge_attr  = (const float*)d_in[2];
    // d_in[3] rev_edge_index: by construction rev[e] == e^1, folded into kernels
    const float* W_i = (const float*)d_in[4];
    const float* b_i = (const float*)d_in[5];
    const float* W_h = (const float*)d_in[6];
    const float* b_h = (const float*)d_in[7];
    const float* W_o = (const float*)d_in[8];
    const float* b_o = (const float*)d_in[9];
    float* out = (float*)d_out;

    const int E = in_sizes[1] / 2;   // 400000
    const int N = in_sizes[0] / 64;  // 400000

    const size_t hbytes = (size_t)E * HID * sizeof(float);
    char* ws = (char*)d_ws;
    float* h0 = (float*)(ws);
    float* h  = (float*)(ws + hbytes);
    float* m  = (float*)(ws + 2 * hbytes);
    // bf16 weight buffers (col-major, padded stride KPS = KP+8), 256B-aligned
    size_t woff = 3 * hbytes;
    bf16_t* Bw_i = (bf16_t*)(ws + woff);  woff += ((size_t)128 * 104 * 2 + 255) & ~(size_t)255;
    bf16_t* Bw_h = (bf16_t*)(ws + woff);  woff += ((size_t)128 * 136 * 2 + 255) & ~(size_t)255;
    bf16_t* Bw_o = (bf16_t*)(ws + woff);

    const int gemm_blocks    = (E + 127) / 128;   // 3125
    const int scatter_blocks = (E + 7) / 8;       // 50000

    // one-time (per call) weight conversion: f32 [K,128] -> bf16 col-major padded
    convert_weights_kernel<<<(128 * 104 + 255) / 256, 256, 0, stream>>>(W_i, Bw_i,  80, 104);
    convert_weights_kernel<<<(128 * 136 + 255) / 256, 256, 0, stream>>>(W_h, Bw_h, 128, 136);
    convert_weights_kernel<<<(128 * 200 + 255) / 256, 256, 0, stream>>>(W_o, Bw_o, 192, 200);

    // h0 = relu([x[src], edge_attr] @ W_i + b_i)
    gemm_wmma_kernel<80, 96, 0><<<gemm_blocks, 256, 0, stream>>>(
        x, edge_attr, edge_index, Bw_i, b_i, nullptr, h0, E);

    const float* hc = h0;
    for (int it = 1; it < 5; ++it) {
        hipMemsetAsync(m, 0, hbytes, stream);
        scatter_kernel<<<scatter_blocks, 256, 0, stream>>>(hc, edge_index, m, E, 1);
        gemm_wmma_kernel<128, 128, 1><<<gemm_blocks, 256, 0, stream>>>(
            m, nullptr, nullptr, Bw_h, b_h, h0, h, E);
        hc = h;
    }

    // m_final = scatter(dst, +h); out = relu(sanitize([x, m_final]) @ W_o + b_o)
    hipMemsetAsync(m, 0, hbytes, stream);
    scatter_kernel<<<scatter_blocks, 256, 0, stream>>>(hc, edge_index, m, E, 0);
    gemm_wmma_kernel<192, 192, 2><<<gemm_blocks, 256, 0, stream>>>(
        x, m, nullptr, Bw_o, b_o, nullptr, out, N);
}
